// BiGRU_6305011991049
// MI455X (gfx1250) — matrix operand order repaired and checked by reading, not compiled, not run on hardware
//
#include <hip/hip_runtime.h>
#include <hip/hip_bf16.h>

// ---------------------------------------------------------------------------
// BiGRU (2-layer, bidirectional, H=512, T=21, B=2048) for gfx1250 / MI455X.
// - Each block owns 32 batch rows and runs the full time loop with h resident
//   in LDS (double-buffered bf16); no inter-block sync needed.
// - All matmuls via v_wmma_f32_16x16x32_bf16; weights pre-packed into the
//   per-lane WMMA B layout for contiguous b128 loads.
// - Both row tiles are fused per column-tile pass (8 live accumulators) to
//   halve L2 weight traffic and double WMMA-per-load density.
// - Forward and backward directions run concurrently via blockIdx.y.
// ---------------------------------------------------------------------------

#define H   512
#define TT_ 21
#define BB_ 2048
#define RB  32            // batch rows per block
#define HSTR 520          // LDS row stride in bf16 elems (multiple of 8)

typedef __attribute__((ext_vector_type(16))) __bf16 v16bf;
typedef __attribute__((ext_vector_type(8)))  float  v8f;

union ABu { uint4 q[2]; v16bf v; };

__device__ __forceinline__ unsigned short f2bf(float f) {
  unsigned u = __builtin_bit_cast(unsigned, f);
  unsigned r = (u + 0x7FFFu + ((u >> 16) & 1u)) >> 16;   // RNE
  return (unsigned short)r;
}
__device__ __forceinline__ float bf2f(unsigned short s) {
  unsigned u = ((unsigned)s) << 16;
  return __builtin_bit_cast(float, u);
}

// ---------------------------------------------------------------------------
// Pack fp32 weight matrix W[1536][srcCols] (torch [3H, K] layout) into the
// WMMA B-matrix per-lane layout, bf16:
//   packed[((ct*kTiles + kt)*32 + lane)*16 + e] =
//       bf16( W[ct*16 + (lane&15)][kt*32 + (lane>>4)*16 + e] )
// so each lane's 16 elements are one contiguous 32-byte chunk.
// ---------------------------------------------------------------------------
__global__ void pack_w_kernel(const float* __restrict__ src,
                              unsigned short* __restrict__ dst,
                              int kTiles, int srcCols) {
  int idx = blockIdx.x * blockDim.x + threadIdx.x;   // one per (ct,kt,lane)
  int total = 96 * kTiles * 32;
  if (idx >= total) return;
  int lane = idx & 31;
  int kt   = (idx >> 5) % kTiles;
  int ct   = idx / (32 * kTiles);
  int g    = ct * 16 + (lane & 15);
  int kbase = kt * 32 + (lane >> 4) * 16;
  unsigned short tmp[16];
#pragma unroll
  for (int e = 0; e < 16; ++e)
    tmp[e] = f2bf(src[(size_t)g * srcCols + kbase + e]);
  uint4* d = (uint4*)&dst[(size_t)idx * 16];
  d[0] = *(uint4*)&tmp[0];
  d[1] = *(uint4*)&tmp[8];
}

// ---------------------------------------------------------------------------
// One GRU layer, both directions (blockIdx.y = 0 fwd / 1 bwd).
// Block = 256 threads (8 wave32), owns RB=32 batch rows, loops all T steps.
// LAYER 0: input dim 2 handled scalarly in the gate phase.
// LAYER 1: input projection (K=1024) fused into the WMMA K-loop, reading the
//          previous layer's bf16 output directly from global.
// ---------------------------------------------------------------------------
template <int LAYER>
__global__ __launch_bounds__(256) void gru_kernel(
    const float* __restrict__ x0,              // layer0: X [B][T][2] fp32
    const unsigned short* __restrict__ ain,    // layer1: prev out [B][T][1024] bf16
    const unsigned short* __restrict__ pwhh_f,
    const unsigned short* __restrict__ pwhh_b, // packed W_hh [96][16][32][16]
    const unsigned short* __restrict__ pwih_f,
    const unsigned short* __restrict__ pwih_b, // layer1 packed W_ih [96][32][32][16]
    const float* __restrict__ wih0_f,
    const float* __restrict__ wih0_b,          // layer0 W_ih [1536][2] fp32
    const float* __restrict__ bih_f, const float* __restrict__ bih_b,
    const float* __restrict__ bhh_f, const float* __restrict__ bhh_b,
    unsigned short* __restrict__ out) {        // [B][T][1024] bf16
  __shared__ __align__(16) unsigned short hsh[2][RB * HSTR];
  __shared__ float xsh[RB * 2];

  const int dir = blockIdx.y;
  const unsigned short* pwhh = dir ? pwhh_b : pwhh_f;
  const unsigned short* pwih = dir ? pwih_b : pwih_f;
  const float* wih0 = dir ? wih0_b : wih0_f;
  const float* b_ih = dir ? bih_b : bih_f;
  const float* b_hh = dir ? bhh_b : bhh_f;
  const int dirOff = dir * 512;

  const int tid   = threadIdx.x;
  const int lane  = tid & 31;
  const int wave  = tid >> 5;
  const int half  = lane >> 4;
  const int nidx  = lane & 15;
  const int bRow0 = blockIdx.x * RB;

  // h0 = 0
  for (int i = tid; i < RB * HSTR; i += 256) hsh[0][i] = 0;
  __syncthreads();

  int p = 0;
  for (int s = 0; s < TT_; ++s) {
    const int tt = dir ? (TT_ - 1 - s) : s;

    if (LAYER == 0) {
      if (tid < RB * 2)
        xsh[tid] = x0[((size_t)(bRow0 + (tid >> 1)) * TT_ + tt) * 2 + (tid & 1)];
      __syncthreads();
    }

    // A-source row bases for the two fused row tiles (this lane's A row)
    const unsigned short* hrow[2];
    const unsigned short* arow[2];
#pragma unroll
    for (int r = 0; r < 2; ++r) {
      const int am = r * 16 + nidx;
      hrow[r] = &hsh[p][am * HSTR];
      arow[r] = (LAYER == 1)
                    ? (ain + ((size_t)(bRow0 + am) * TT_ + tt) * 1024)
                    : nullptr;
    }

    for (int c = 0; c < 4; ++c) {
      const int ct = wave * 4 + c;          // r-gate column tile (0..31)
      v8f accR[2] = {v8f{}, v8f{}};
      v8f accZ[2] = {v8f{}, v8f{}};
      v8f accNh[2] = {v8f{}, v8f{}};
      v8f accNx[2] = {v8f{}, v8f{}};

      // ---- hidden-state part: hp = h @ W_hh^T, K = 512 ----
      for (int kt = 0; kt < 16; ++kt) {
        const uint4* br = (const uint4*)(pwhh + (((size_t)ct * 16 + kt) * 32 + lane) * 16);
        const uint4* bz = (const uint4*)(pwhh + (((size_t)(ct + 32) * 16 + kt) * 32 + lane) * 16);
        const uint4* bn = (const uint4*)(pwhh + (((size_t)(ct + 64) * 16 + kt) * 32 + lane) * 16);
        ABu b1, b2, b3;
        b1.q[0] = br[0]; b1.q[1] = br[1];
        b2.q[0] = bz[0]; b2.q[1] = bz[1];
        b3.q[0] = bn[0]; b3.q[1] = bn[1];
#pragma unroll
        for (int r = 0; r < 2; ++r) {
          ABu a;
          const unsigned short* hp = hrow[r] + kt * 32 + half * 16;
          a.q[0] = *(const uint4*)hp;          // K = half*16 + 0..7
          a.q[1] = *(const uint4*)(hp + 8);    // K = half*16 + 8..15
          accR[r]  = __builtin_amdgcn_wmma_f32_16x16x32_bf16(false, a.v, false, b1.v, (short)0, accR[r],  false, false);
          accZ[r]  = __builtin_amdgcn_wmma_f32_16x16x32_bf16(false, a.v, false, b2.v, (short)0, accZ[r],  false, false);
          accNh[r] = __builtin_amdgcn_wmma_f32_16x16x32_bf16(false, a.v, false, b3.v, (short)0, accNh[r], false, false);
        }
      }

      // ---- input part (layer 1 only): xp = x_t @ W_ih^T, K = 1024 ----
      if (LAYER == 1) {
        for (int kt = 0; kt < 32; ++kt) {
          const uint4* br = (const uint4*)(pwih + (((size_t)ct * 32 + kt) * 32 + lane) * 16);
          const uint4* bz = (const uint4*)(pwih + (((size_t)(ct + 32) * 32 + kt) * 32 + lane) * 16);
          const uint4* bn = (const uint4*)(pwih + (((size_t)(ct + 64) * 32 + kt) * 32 + lane) * 16);
          ABu b1, b2, b3;
          b1.q[0] = br[0]; b1.q[1] = br[1];
          b2.q[0] = bz[0]; b2.q[1] = bz[1];
          b3.q[0] = bn[0]; b3.q[1] = bn[1];
#pragma unroll
          for (int r = 0; r < 2; ++r) {
            ABu a;
            const unsigned short* ap = arow[r] + kt * 32 + half * 16;
            a.q[0] = *(const uint4*)ap;
            a.q[1] = *(const uint4*)(ap + 8);
            accR[r]  = __builtin_amdgcn_wmma_f32_16x16x32_bf16(false, a.v, false, b1.v, (short)0, accR[r],  false, false);
            accZ[r]  = __builtin_amdgcn_wmma_f32_16x16x32_bf16(false, a.v, false, b2.v, (short)0, accZ[r],  false, false);
            accNx[r] = __builtin_amdgcn_wmma_f32_16x16x32_bf16(false, a.v, false, b3.v, (short)0, accNx[r], false, false);
          }
        }
      }

      // ---- gate math in registers (C/D layout: M = v + 8*half, N = nidx) ----
      const int g = ct * 16 + nidx;      // hidden index in [0,512)
      const float bihR = b_ih[g],         bhhR = b_hh[g];
      const float bihZ = b_ih[H + g],     bhhZ = b_hh[H + g];
      const float bihN = b_ih[2 * H + g], bhhN = b_hh[2 * H + g];
      float wr0 = 0.f, wr1 = 0.f, wz0 = 0.f, wz1 = 0.f, wn0 = 0.f, wn1 = 0.f;
      if (LAYER == 0) {
        wr0 = wih0[2 * g];             wr1 = wih0[2 * g + 1];
        wz0 = wih0[2 * (H + g)];       wz1 = wih0[2 * (H + g) + 1];
        wn0 = wih0[2 * (2 * H + g)];   wn1 = wih0[2 * (2 * H + g) + 1];
      }
#pragma unroll
      for (int r = 0; r < 2; ++r) {
#pragma unroll
        for (int v = 0; v < 8; ++v) {
          const int mo = r * 16 + v + half * 8;   // batch row within block
          float xR = 0.f, xZ = 0.f, xN = 0.f;
          if (LAYER == 0) {
            const float xa = xsh[mo * 2], xb = xsh[mo * 2 + 1];
            xR = xa * wr0 + xb * wr1;
            xZ = xa * wz0 + xb * wz1;
            xN = xa * wn0 + xb * wn1;
          }
          const float rr = 1.f / (1.f + __expf(-(accR[r][v] + xR + bihR + bhhR)));
          const float zz = 1.f / (1.f + __expf(-(accZ[r][v] + xZ + bihZ + bhhZ)));
          const float ng = tanhf((accNx[r][v] + xN + bihN) + rr * (accNh[r][v] + bhhN));
          const float hold = bf2f(hsh[p][mo * HSTR + g]);
          const float hnew = (1.f - zz) * ng + zz * hold;
          const unsigned short hb = f2bf(hnew);
          hsh[p ^ 1][mo * HSTR + g] = hb;
          out[((size_t)(bRow0 + mo) * TT_ + tt) * 1024 + dirOff + g] = hb;
        }
      }
    }
    __syncthreads();
    p ^= 1;
  }
}

// ---------------------------------------------------------------------------
// Final linear [2 x 21504] + 2-class softmax. One block per batch row.
// d_out layout: [0:4096) logits, [4096:8192) softmax (tuple concat).
// ---------------------------------------------------------------------------
__global__ __launch_bounds__(256) void linear_softmax_kernel(
    const unsigned short* __restrict__ h2,   // [B][T][1024] bf16
    const float* __restrict__ wlin,          // [2][21504] fp32
    const float* __restrict__ blin,          // [2] fp32
    float* __restrict__ outp) {
  const int b   = blockIdx.x;
  const int tid = threadIdx.x;
  const int NF  = TT_ * 1024;                // 21504
  const unsigned short* row = h2 + (size_t)b * NF;
  float s0 = 0.f, s1 = 0.f;
  for (int i = tid; i < NF; i += 256) {
    const float v = bf2f(row[i]);
    s0 += v * wlin[i];
    s1 += v * wlin[NF + i];
  }
  __shared__ float r0[256], r1[256];
  r0[tid] = s0; r1[tid] = s1;
  __syncthreads();
  for (int off = 128; off > 0; off >>= 1) {
    if (tid < off) { r0[tid] += r0[tid + off]; r1[tid] += r1[tid + off]; }
    __syncthreads();
  }
  if (tid == 0) {
    const float l0 = r0[0] + blin[0];
    const float l1 = r1[0] + blin[1];
    outp[b * 2 + 0] = l0;
    outp[b * 2 + 1] = l1;
    const float m  = fmaxf(l0, l1);
    const float e0 = __expf(l0 - m), e1 = __expf(l1 - m);
    const float inv = 1.f / (e0 + e1);
    outp[BB_ * 2 + b * 2 + 0] = e0 * inv;
    outp[BB_ * 2 + b * 2 + 1] = e1 * inv;
  }
}

// ---------------------------------------------------------------------------
extern "C" void kernel_launch(void* const* d_in, const int* in_sizes, int n_in,
                              void* d_out, int out_size, void* d_ws, size_t ws_size,
                              hipStream_t stream) {
  (void)in_sizes; (void)n_in; (void)out_size; (void)ws_size;

  const float* X        = (const float*)d_in[0];
  const float* w_ih_l0f = (const float*)d_in[1];
  const float* w_hh_l0f = (const float*)d_in[2];
  const float* b_ih_l0f = (const float*)d_in[3];
  const float* b_hh_l0f = (const float*)d_in[4];
  const float* w_ih_l0b = (const float*)d_in[5];
  const float* w_hh_l0b = (const float*)d_in[6];
  const float* b_ih_l0b = (const float*)d_in[7];
  const float* b_hh_l0b = (const float*)d_in[8];
  const float* w_ih_l1f = (const float*)d_in[9];
  const float* w_hh_l1f = (const float*)d_in[10];
  const float* b_ih_l1f = (const float*)d_in[11];
  const float* b_hh_l1f = (const float*)d_in[12];
  const float* w_ih_l1b = (const float*)d_in[13];
  const float* w_hh_l1b = (const float*)d_in[14];
  const float* b_ih_l1b = (const float*)d_in[15];
  const float* b_hh_l1b = (const float*)d_in[16];
  const float* w_lin    = (const float*)d_in[17];
  const float* b_lin    = (const float*)d_in[18];

  // ---- workspace layout ----
  unsigned char* ws = (unsigned char*)d_ws;
  const size_t PW_HH = (size_t)96 * 16 * 32 * 16 * sizeof(unsigned short); // 1.5 MB
  const size_t PW_IH = (size_t)96 * 32 * 32 * 16 * sizeof(unsigned short); // 3 MB
  unsigned short* pwhh_l0f = (unsigned short*)(ws);
  unsigned short* pwhh_l0b = (unsigned short*)(ws + PW_HH);
  unsigned short* pwhh_l1f = (unsigned short*)(ws + 2 * PW_HH);
  unsigned short* pwhh_l1b = (unsigned short*)(ws + 3 * PW_HH);
  unsigned short* pwih_l1f = (unsigned short*)(ws + 4 * PW_HH);
  unsigned short* pwih_l1b = (unsigned short*)(ws + 4 * PW_HH + PW_IH);
  unsigned short* h1 = (unsigned short*)(ws + 4 * PW_HH + 2 * PW_IH);
  unsigned short* h2 = h1 + (size_t)BB_ * TT_ * 1024;

  // ---- pack weights into WMMA B-matrix layout ----
  {
    const int tot_hh = 96 * 16 * 32, tot_ih = 96 * 32 * 32;
    dim3 blk(256);
    pack_w_kernel<<<(tot_hh + 255) / 256, blk, 0, stream>>>(w_hh_l0f, pwhh_l0f, 16, 512);
    pack_w_kernel<<<(tot_hh + 255) / 256, blk, 0, stream>>>(w_hh_l0b, pwhh_l0b, 16, 512);
    pack_w_kernel<<<(tot_hh + 255) / 256, blk, 0, stream>>>(w_hh_l1f, pwhh_l1f, 16, 512);
    pack_w_kernel<<<(tot_hh + 255) / 256, blk, 0, stream>>>(w_hh_l1b, pwhh_l1b, 16, 512);
    pack_w_kernel<<<(tot_ih + 255) / 256, blk, 0, stream>>>(w_ih_l1f, pwih_l1f, 32, 1024);
    pack_w_kernel<<<(tot_ih + 255) / 256, blk, 0, stream>>>(w_ih_l1b, pwih_l1b, 32, 1024);
  }

  // ---- layer 0: fwd+bwd concurrently (blockIdx.y), writes h1 ----
  const dim3 grid(BB_ / RB, 2);   // 64 x 2 blocks
  gru_kernel<0><<<grid, 256, 0, stream>>>(
      X, nullptr, pwhh_l0f, pwhh_l0b, nullptr, nullptr,
      w_ih_l0f, w_ih_l0b, b_ih_l0f, b_ih_l0b, b_hh_l0f, b_hh_l0b, h1);

  // ---- layer 1: fwd+bwd concurrently, reads h1, writes h2 ----
  gru_kernel<1><<<grid, 256, 0, stream>>>(
      nullptr, h1, pwhh_l1f, pwhh_l1b, pwih_l1f, pwih_l1b,
      nullptr, nullptr, b_ih_l1f, b_ih_l1b, b_hh_l1f, b_hh_l1b, h2);

  // ---- linear + softmax ----
  linear_softmax_kernel<<<BB_, 256, 0, stream>>>(h2, w_lin, b_lin, (float*)d_out);
}
